// Block_62904091018073
// MI455X (gfx1250) — matrix operand order repaired and checked by reading, not compiled, not run on hardware
//
#include <hip/hip_runtime.h>
#include <hip/hip_bf16.h>
#include <math.h>

#define B_   2
#define T_   1024
#define C_   1024
#define H_   16
#define HS_  64
#define ROT_ 32
#define FFN_ 3072
#define BT_  (B_ * T_)
#define LN_EPS_ 1e-5f

typedef __attribute__((ext_vector_type(16))) _Float16 v16h;
typedef __attribute__((ext_vector_type(8)))  _Float16 v8h;
typedef __attribute__((ext_vector_type(8)))  float    v8f;

// ---------------- f32 -> f16 convert ----------------
__global__ void k_f32_to_f16(const float* __restrict__ src, _Float16* __restrict__ dst, int n) {
  int i = (blockIdx.x * blockDim.x + threadIdx.x) * 4;
  if (i + 3 < n) {
    float4 v = *(const float4*)(src + i);
    dst[i + 0] = (_Float16)v.x; dst[i + 1] = (_Float16)v.y;
    dst[i + 2] = (_Float16)v.z; dst[i + 3] = (_Float16)v.w;
  } else {
    for (int j = i; j < n; ++j) dst[j] = (_Float16)src[j];
  }
}

// ---------------- LN1 + time_shift_half -> f16 ----------------
__global__ __launch_bounds__(256) void k_ln_shift(
    const float* __restrict__ x, const float* __restrict__ w, const float* __restrict__ b,
    _Float16* __restrict__ xs) {
  int row = blockIdx.x;            // b*T + t
  int t = row & (T_ - 1);
  int tid = threadIdx.x;
  __shared__ float s1[256], s2[256];
  __shared__ float stats[4];
  const float* cur = x + (size_t)row * C_;
  const float* prev = cur - C_;
  float a = 0.f, q = 0.f;
  for (int c = tid; c < C_; c += 256) { float v = cur[c]; a += v; q += v * v; }
  s1[tid] = a; s2[tid] = q; __syncthreads();
  for (int s = 128; s > 0; s >>= 1) {
    if (tid < s) { s1[tid] += s1[tid + s]; s2[tid] += s2[tid + s]; } __syncthreads();
  }
  if (tid == 0) {
    float m = s1[0] / C_; float var = s2[0] / C_ - m * m;
    stats[0] = m; stats[1] = rsqrtf(var + LN_EPS_);
  }
  __syncthreads();
  a = 0.f; q = 0.f;
  if (t > 0) for (int c = tid; c < C_; c += 256) { float v = prev[c]; a += v; q += v * v; }
  s1[tid] = a; s2[tid] = q; __syncthreads();
  for (int s = 128; s > 0; s >>= 1) {
    if (tid < s) { s1[tid] += s1[tid + s]; s2[tid] += s2[tid + s]; } __syncthreads();
  }
  if (tid == 0) {
    float m = s1[0] / C_; float var = s2[0] / C_ - m * m;
    stats[2] = m; stats[3] = rsqrtf(var + LN_EPS_);
  }
  __syncthreads();
  float mc = stats[0], rc = stats[1], mp = stats[2], rp = stats[3];
  _Float16* orow = xs + (size_t)row * C_;
  for (int c = tid; c < C_; c += 256) {
    float v;
    if (c < C_ / 2) v = (t == 0) ? 0.f : (prev[c] - mp) * rp * w[c] + b[c];
    else            v = (cur[c] - mc) * rc * w[c] + b[c];
    orow[c] = (_Float16)v;
  }
}

// ---------------- plain LN -> f16 ----------------
__global__ __launch_bounds__(256) void k_ln(
    const float* __restrict__ x, const float* __restrict__ w, const float* __restrict__ b,
    _Float16* __restrict__ out) {
  int row = blockIdx.x;
  int tid = threadIdx.x;
  __shared__ float s1[256], s2[256];
  const float* r = x + (size_t)row * C_;
  float a = 0.f, q = 0.f;
  for (int c = tid; c < C_; c += 256) { float v = r[c]; a += v; q += v * v; }
  s1[tid] = a; s2[tid] = q; __syncthreads();
  for (int s = 128; s > 0; s >>= 1) {
    if (tid < s) { s1[tid] += s1[tid + s]; s2[tid] += s2[tid + s]; } __syncthreads();
  }
  float m = s1[0] / C_; float var = s2[0] / C_ - m * m;
  float rs = rsqrtf(var + LN_EPS_);
  _Float16* o = out + (size_t)row * C_;
  for (int c = tid; c < C_; c += 256) o[c] = (_Float16)((r[c] - m) * rs * w[c] + b[c]);
}

// ---------------- generic tiled WMMA GEMM (ping-pong double buffered) ----------------
// C[M,N] = alpha * (A[M,K] @ B[K,N]) (+bias[n]) (*gamma[m%gammaMod]) (+res[m,n])
// block tile 64x128, 8 waves (wave32), wave tile 32x32 (2x2 of 16x16), K-step 32.
__global__ __launch_bounds__(256) void k_gemm_wmma(
    const _Float16* __restrict__ A, const _Float16* __restrict__ Bm,
    const float* __restrict__ bias, const float* __restrict__ gamma, int gammaMod,
    const float* __restrict__ res, void* __restrict__ Out, int outF16,
    int M, int N, int K, int lda, int ldb, int ldc, float alpha,
    long long sA, long long sB, long long sC, int batchInner,
    long long sA2, long long sB2, long long sC2) {
  __shared__ _Float16 As[2][64][40];    // [buf][m][k], +8 pad
  __shared__ _Float16 Bs[2][128][40];   // [buf][n][k], +8 pad
  int z  = blockIdx.z;
  int zo = z / batchInner, zi = z % batchInner;
  A  += (long long)zo * sA2 + (long long)zi * sA;
  Bm += (long long)zo * sB2 + (long long)zi * sB;
  long long coff = (long long)zo * sC2 + (long long)zi * sC;
  if (res) res += coff;
  int tid = threadIdx.x, lane = tid & 31, wave = tid >> 5;
  int wm = wave & 1, wn = wave >> 1;
  int l15 = lane & 15, hs = lane >> 4;
  int blockM = blockIdx.y * 64, blockN = blockIdx.x * 128;
  v8f acc[2][2] = {};
  // A loader: 64 rows x 4 segments of 8 halves
  int arow = tid >> 2, aseg = (tid & 3) << 3;
  int gmA = blockM + arow;
  // B loader: k-pairs x 8 n columns (paired-dword transpose stores)
  int bkk = (tid >> 4) << 1;            // 0,2,..,30
  int bn0 = (tid & 15) << 3;            // 0,8,..,120
  bool bOk = (blockN + bn0) < N;

  v8h av = {}, blo = {}, bhi = {};
  // prime tile 0 into registers
  if (gmA < M) av = *(const v8h*)(A + (long long)gmA * lda + aseg);
  if (bOk) {
    const _Float16* bp = Bm + (long long)bkk * ldb + (blockN + bn0);
    blo = *(const v8h*)bp;
    bhi = *(const v8h*)(bp + ldb);
  }

  int buf = 0;
  for (int k0 = 0; k0 < K; k0 += 32) {
    // ---- commit staged registers to LDS[buf] ----
    *(v8h*)&As[buf][arow][aseg] = av;
#pragma unroll
    for (int j = 0; j < 8; ++j) {
      union { _Float16 h[2]; unsigned u; } pk;
      pk.h[0] = blo[j]; pk.h[1] = bhi[j];
      *(unsigned*)&Bs[buf][bn0 + j][bkk] = pk.u;   // {B[k][n], B[k+1][n]}
    }
    __syncthreads();

    // ---- prefetch next tile into registers (overlaps with WMMA below) ----
    if (k0 + 32 < K) {
      av = {}; blo = {}; bhi = {};
      if (gmA < M) {
        const _Float16* ap = A + (long long)gmA * lda + (k0 + 32 + aseg);
        av = *(const v8h*)ap;
        if (k0 + 64 < K) __builtin_prefetch(ap + 32, 0, 1);
      }
      if (bOk) {
        const _Float16* bp = Bm + (long long)(k0 + 32 + bkk) * ldb + (blockN + bn0);
        blo = *(const v8h*)bp;
        bhi = *(const v8h*)(bp + ldb);
        if (k0 + 64 < K) __builtin_prefetch(bp + 32LL * ldb, 0, 1);
      }
    }

    // ---- fragments in contiguous k order (k = 16*half + s, same for A and B) ----
    union { v16h v; v8h h[2]; } afr[2], bfr[2];
#pragma unroll
    for (int mf = 0; mf < 2; ++mf) {
      int m = wm * 32 + mf * 16 + l15;
      int ab = hs * 16;                              // lanes 16-31 hold K=16..31
      afr[mf].h[0] = *(const v8h*)&As[buf][m][ab];        // K = ab .. ab+7
      afr[mf].h[1] = *(const v8h*)&As[buf][m][ab + 8];    // K = ab+8 .. ab+15
    }
#pragma unroll
    for (int nf = 0; nf < 2; ++nf) {
      int n = wn * 32 + nf * 16 + l15;
      int kb = hs * 16;                              // lanes 16-31 hold K=16..31
      bfr[nf].h[0] = *(const v8h*)&Bs[buf][n][kb];
      bfr[nf].h[1] = *(const v8h*)&Bs[buf][n][kb + 8];
    }
#pragma unroll
    for (int mf = 0; mf < 2; ++mf)
#pragma unroll
      for (int nf = 0; nf < 2; ++nf)
        acc[mf][nf] = __builtin_amdgcn_wmma_f32_16x16x32_f16(
            false, afr[mf].v, false, bfr[nf].v, (short)0, acc[mf][nf], false, false);
    buf ^= 1;
    // single barrier per step: reads of buf are complete (dscnt waited by wmma
    // consumption) before the *next* iteration's barrier, so writing buf again
    // two iterations later is safe.
  }

  // ---- epilogue ----
#pragma unroll
  for (int mf = 0; mf < 2; ++mf)
#pragma unroll
    for (int nf = 0; nf < 2; ++nf) {
      int gn = blockN + wn * 32 + nf * 16 + l15;
      if (gn >= N) continue;
      float bvv = bias ? bias[gn] : 0.f;
#pragma unroll
      for (int r = 0; r < 8; ++r) {
        int gm = blockM + wm * 32 + mf * 16 + r + hs * 8;
        if (gm >= M) continue;
        float v = acc[mf][nf][r] * alpha + bvv;
        if (gamma) v *= gamma[gm % gammaMod];
        if (res) v += res[(long long)gm * ldc + gn];
        long long oi = coff + (long long)gm * ldc + gn;
        if (outF16) ((_Float16*)Out)[oi] = (_Float16)v;
        else        ((float*)Out)[oi] = v;
      }
    }
}

// ---------------- rotary + pack q/kT/v to f16 per-head ----------------
__global__ void k_rotary_pack(const float* __restrict__ qf, const float* __restrict__ kf,
                              const float* __restrict__ vf,
                              _Float16* __restrict__ q16, _Float16* __restrict__ k16T,
                              _Float16* __restrict__ v16) {
  int idx = blockIdx.x * blockDim.x + threadIdx.x;   // < B*H*T*HS = 2^21
  int d = idx & 63;
  int t = (idx >> 6) & 1023;
  int h = (idx >> 16) & 15;
  int b = idx >> 20;
  size_t src = ((size_t)(b * T_ + t)) * C_ + h * HS_ + d;
  float qv = qf[src], kv = kf[src], vv = vf[src];
  if (d < ROT_) {
    int j = d & 15;
    float inv = __expf(-0.5756462732f * (float)j);   // 10000^(-j/16)
    float ang = (float)t * inv;
    float sn, cs;
    __sincosf(ang, &sn, &cs);
    if (d < 16) {
      float qp = qf[src + 16], kp = kf[src + 16];
      qv = qv * cs - qp * sn; kv = kv * cs - kp * sn;
    } else {
      float qp = qf[src - 16], kp = kf[src - 16];
      qv = qv * cs + qp * sn; kv = kv * cs + kp * sn;
    }
  }
  size_t bh = (size_t)(b * H_ + h);
  q16 [bh * T_ * HS_ + (size_t)t * HS_ + d] = (_Float16)qv;
  v16 [bh * T_ * HS_ + (size_t)t * HS_ + d] = (_Float16)vv;
  k16T[bh * T_ * HS_ + (size_t)d * T_  + t] = (_Float16)kv;
}

// ---------------- causal softmax + decay, in place on f16 scores ----------------
__global__ __launch_bounds__(256) void k_softmax_decay(
    _Float16* __restrict__ att, const float* __restrict__ tw,
    const float* __restrict__ ta, const float* __restrict__ tb) {
  int row = blockIdx.x;            // ((b*H)+h)*T + q
  int q = row & (T_ - 1);
  int h = (row >> 10) & (H_ - 1);
  int tid = threadIdx.x;
  _Float16* s = att + (size_t)row * T_;
  __shared__ float red[256];
  float mx = -3.0e38f;
  for (int k = tid; k <= q; k += 256) mx = fmaxf(mx, (float)s[k]);
  red[tid] = mx; __syncthreads();
  for (int st = 128; st > 0; st >>= 1) {
    if (tid < st) red[tid] = fmaxf(red[tid], red[tid + st]); __syncthreads();
  }
  mx = red[0]; __syncthreads();
  float sum = 0.f;
  for (int k = tid; k <= q; k += 256) sum += __expf((float)s[k] - mx);
  red[tid] = sum; __syncthreads();
  for (int st = 128; st > 0; st >>= 1) {
    if (tid < st) red[tid] += red[tid + st]; __syncthreads();
  }
  float inv = 1.f / red[0];
  float be = tb[h * T_ + q];
  for (int k = tid; k < T_; k += 256) {
    float o = 0.f;
    if (k <= q) {
      float e = __expf((float)s[k] - mx) * inv;
      float w = tw[h * T_ + (T_ - 1 + k - q)] * ta[h * T_ + k] * be;
      o = e * w;
    }
    s[k] = (_Float16)o;
  }
}

// ---------------- 16x16 head mix, in place ----------------
__global__ __launch_bounds__(256) void k_mix(_Float16* __restrict__ att,
                                             const float* __restrict__ mixw) {
  __shared__ float mw[H_ * H_];
  int tid = threadIdx.x;
  if (tid < H_ * H_) mw[tid] = mixw[tid];
  __syncthreads();
  long long idx = (long long)blockIdx.x * 256 + tid;   // < B*T*T
  int b = (int)(idx >> 20);                            // T*T = 2^20
  long long rem = idx & ((1LL << 20) - 1);
  _Float16* p = att + (((long long)b * H_) << 20) + rem;
  float a[H_];
#pragma unroll
  for (int i = 0; i < H_; ++i) a[i] = (float)p[(long long)i << 20];
#pragma unroll
  for (int o = 0; o < H_; ++o) {
    float v = 0.f;
#pragma unroll
    for (int i = 0; i < H_; ++i) v += mw[o * H_ + i] * a[i];
    p[(long long)o << 20] = (_Float16)v;
  }
}

// ---------------- geglu gate: g = f16(gelu(kk) * vv) ----------------
__global__ void k_geglu(const float* __restrict__ kk, const float* __restrict__ vv,
                        _Float16* __restrict__ g, int n) {
  int i = blockIdx.x * blockDim.x + threadIdx.x;
  if (i >= n) return;
  float x = kk[i];
  float ge = 0.5f * x * (1.f + erff(x * 0.70710678f));
  g[i] = (_Float16)(ge * vv[i]);
}

// ---------------- host side ----------------
extern "C" void kernel_launch(void* const* d_in, const int* in_sizes, int n_in,
                              void* d_out, int out_size, void* d_ws, size_t ws_size,
                              hipStream_t stream) {
  (void)in_sizes; (void)n_in; (void)out_size; (void)ws_size;
  const float* x    = (const float*)d_in[0];
  const float* ln1w = (const float*)d_in[1];
  const float* ln1b = (const float*)d_in[2];
  const float* ln2w = (const float*)d_in[3];
  const float* ln2b = (const float*)d_in[4];
  const float* Wq   = (const float*)d_in[5];
  const float* bq   = (const float*)d_in[6];
  const float* Wk   = (const float*)d_in[7];
  const float* bk   = (const float*)d_in[8];
  const float* Wv   = (const float*)d_in[9];
  const float* bv   = (const float*)d_in[10];
  const float* Wo   = (const float*)d_in[11];
  const float* bo   = (const float*)d_in[12];
  const float* tw   = (const float*)d_in[13];
  const float* ta   = (const float*)d_in[14];
  const float* tb   = (const float*)d_in[15];
  const float* tg   = (const float*)d_in[16];
  const float* mixw = (const float*)d_in[17];
  const float* Wkg  = (const float*)d_in[18];
  const float* bkg  = (const float*)d_in[19];
  const float* Wvg  = (const float*)d_in[20];
  const float* bvg  = (const float*)d_in[21];
  const float* Wwg  = (const float*)d_in[22];
  const float* bwg  = (const float*)d_in[23];

  char* ws = (char*)d_ws;
  size_t off = 0;
  auto alloc = [&](size_t bytes) { size_t o = off; off += (bytes + 255) & ~(size_t)255; return o; };
  const size_t CC = (size_t)C_ * C_, CF = (size_t)C_ * FFN_;

  _Float16* w16q  = (_Float16*)(ws + alloc(CC * 2));
  _Float16* w16k  = (_Float16*)(ws + alloc(CC * 2));
  _Float16* w16v  = (_Float16*)(ws + alloc(CC * 2));
  _Float16* w16o  = (_Float16*)(ws + alloc(CC * 2));
  _Float16* w16kg = (_Float16*)(ws + alloc(CF * 2));
  _Float16* w16vg = (_Float16*)(ws + alloc(CF * 2));
  _Float16* w16wg = (_Float16*)(ws + alloc(CF * 2));
  _Float16* xs16  = (_Float16*)(ws + alloc((size_t)BT_ * C_ * 2));
  float*    qf    = (float*)   (ws + alloc((size_t)BT_ * C_ * 4));
  float*    kf    = (float*)   (ws + alloc((size_t)BT_ * C_ * 4));
  float*    vf    = (float*)   (ws + alloc((size_t)BT_ * C_ * 4));
  _Float16* q16   = (_Float16*)(ws + alloc((size_t)BT_ * C_ * 2));
  _Float16* k16T  = (_Float16*)(ws + alloc((size_t)BT_ * C_ * 2));
  _Float16* v16   = (_Float16*)(ws + alloc((size_t)BT_ * C_ * 2));
  _Float16* att   = (_Float16*)(ws + alloc((size_t)B_ * H_ * T_ * T_ * 2));
  float*    yf    = (float*)   (ws + alloc((size_t)BT_ * C_ * 4));
  float*    x1    = (float*)   (ws + alloc((size_t)BT_ * C_ * 4));
  // aliases into dead regions
  _Float16* y16  = xs16;                                        // xs dead after QKV
  _Float16* h116 = q16;                                         // q16 dead after scores
  float*    kkf  = (float*)att;                                 // att dead after AV
  float*    vvf  = (float*)((char*)att + (size_t)BT_ * FFN_ * 4);
  _Float16* g16  = (_Float16*)qf;                               // qf/kf dead after rotary

  auto conv = [&](const float* s, _Float16* d, size_t n) {
    k_f32_to_f16<<<(unsigned)((n / 4 + 255) / 256), 256, 0, stream>>>(s, d, (int)n);
  };
  auto gemm = [&](const _Float16* A, const _Float16* Bm, const float* bias,
                  const float* gamma, int gammaMod, const float* res,
                  void* out, int outF16, int M, int N, int K,
                  int lda, int ldb, int ldc, float alpha,
                  long long sA, long long sB, long long sC, int batch, int batchInner,
                  long long sA2, long long sB2, long long sC2) {
    dim3 g((unsigned)((N + 127) / 128), (unsigned)((M + 63) / 64), (unsigned)batch);
    k_gemm_wmma<<<g, 256, 0, stream>>>(A, Bm, bias, gamma, gammaMod, res, out, outF16,
                                       M, N, K, lda, ldb, ldc, alpha,
                                       sA, sB, sC, batchInner, sA2, sB2, sC2);
  };

  // 1) weights to f16
  conv(Wq, w16q, CC);  conv(Wk, w16k, CC);  conv(Wv, w16v, CC);  conv(Wo, w16o, CC);
  conv(Wkg, w16kg, CF); conv(Wvg, w16vg, CF); conv(Wwg, w16wg, CF);

  // 2) LN1 + time-shift
  k_ln_shift<<<BT_, 256, 0, stream>>>(x, ln1w, ln1b, xs16);

  // 3) QKV projections
  gemm(xs16, w16q, bq, nullptr, 1, nullptr, qf, 0, BT_, C_, C_, C_, C_, C_, 1.f,
       0, 0, 0, 1, 1, 0, 0, 0);
  gemm(xs16, w16k, bk, nullptr, 1, nullptr, kf, 0, BT_, C_, C_, C_, C_, C_, 1.f,
       0, 0, 0, 1, 1, 0, 0, 0);
  gemm(xs16, w16v, bv, nullptr, 1, nullptr, vf, 0, BT_, C_, C_, C_, C_, C_, 1.f,
       0, 0, 0, 1, 1, 0, 0, 0);

  // 4) rotary + per-head pack
  k_rotary_pack<<<(B_ * H_ * T_ * HS_) / 256, 256, 0, stream>>>(qf, kf, vf, q16, k16T, v16);

  // 5) scores = 0.125 * q @ kT (batched over b*h), f16 out
  gemm(q16, k16T, nullptr, nullptr, 1, nullptr, att, 1, T_, T_, HS_,
       HS_, T_, T_, 0.125f,
       (long long)T_ * HS_, (long long)HS_ * T_, (long long)T_ * T_,
       B_ * H_, B_ * H_, 0, 0, 0);

  // 6) causal softmax + decay (in place)
  k_softmax_decay<<<B_ * H_ * T_, 256, 0, stream>>>(att, tw, ta, tb);

  // 7) head mix (in place)
  k_mix<<<(B_ * T_ * T_) / 256, 256, 0, stream>>>(att, mixw);

  // 8) y = att @ v, scattered into (b,t,c)
  gemm(att, v16, nullptr, nullptr, 1, nullptr, yf, 0, T_, HS_, T_,
       T_, HS_, C_, 1.f,
       (long long)T_ * T_, (long long)T_ * HS_, (long long)HS_,
       B_ * H_, H_,
       (long long)H_ * T_ * T_, (long long)H_ * T_ * HS_, (long long)T_ * C_);

  // 9) y -> f16; out proj with gamma + residual -> x1
  conv(yf, y16, (size_t)BT_ * C_);
  gemm(y16, w16o, bo, tg, T_, x, x1, 0, BT_, C_, C_, C_, C_, C_, 1.f,
       0, 0, 0, 1, 1, 0, 0, 0);

  // 10) LN2 -> f16
  k_ln<<<BT_, 256, 0, stream>>>(x1, ln2w, ln2b, h116);

  // 11) FFN: kk, vv
  gemm(h116, w16kg, bkg, nullptr, 1, nullptr, kkf, 0, BT_, FFN_, C_, C_, FFN_, FFN_, 1.f,
       0, 0, 0, 1, 1, 0, 0, 0);
  gemm(h116, w16vg, bvg, nullptr, 1, nullptr, vvf, 0, BT_, FFN_, C_, C_, FFN_, FFN_, 1.f,
       0, 0, 0, 1, 1, 0, 0, 0);

  // 12) gate
  k_geglu<<<(BT_ * FFN_ + 255) / 256, 256, 0, stream>>>(kkf, vvf, g16, BT_ * FFN_);

  // 13) final proj + bias + residual -> d_out
  gemm(g16, w16wg, bwg, nullptr, 1, x1, d_out, 0, BT_, C_, FFN_, FFN_, C_, C_, 1.f,
       0, 0, 0, 1, 1, 0, 0, 0);
}
